// MultiHeadAttentionLayer_74663711474279
// MI455X (gfx1250) — matrix operand order repaired and checked by reading, not compiled, not run on hardware
//
#include <hip/hip_runtime.h>
#include <hip/hip_bf16.h>

// ---------------------------------------------------------------------------
// MHA (N=2, L=2048, H=1024, HEADS=16, HD=64), column-axis softmax, causal.
// All GEMMs via V_WMMA_F32_16X16X32_F16 (f32 accumulate).
// Pipeline:
//   K0 x4 : WT = W^T as f16 [n][k]  (one-time, L2-resident; makes every
//           B-fragment two contiguous 16B per-lane reads)
//   K1 x3 : Q = (Hq@Wq+b)*s  -> f16 [bat][i][64]    (s = 1/32, applied to Q,K)
//           K = (Hk@Wk+b)*s  -> f16 [bat][j][64]
//           V = (Hv@Wv+b)    -> f16 transposed [bat][c][j]
//   K2    : per key-column j: m_j = max_{i>=j} S[i,j], s_j = sum exp  (online)
//   K3    : per query tile i: ctx[i,:] = sum_j exp(S[i,j]-m_j)/s_j * V[j,:]
//   K4    : out = ctx @ Wo + bo   (f32 out)
// ---------------------------------------------------------------------------

typedef __attribute__((ext_vector_type(16))) _Float16 v16h;
typedef __attribute__((ext_vector_type(4)))  _Float16 v4h;
typedef __attribute__((ext_vector_type(8)))  float    v8f;

#define CN      2
#define CL      2048
#define CH      1024
#define CHEADS  16
#define CHD     64
#define NBAT    (CN * CHEADS)   // 32
#define MROWS   (CN * CL)       // 4096

__device__ __forceinline__ v8f wmma16(v16h a, v16h b, v8f c) {
  return __builtin_amdgcn_wmma_f32_16x16x32_f16(false, a, false, b, (short)0, c,
                                                false, false);
}

// A-fragment (16x32), lane l: row = l&15, kbase = (l>>4)*8.
// p must point at (row, k0 + kbase); elements are {p[0..7], p[16..23]}.
template <typename TIN>
__device__ __forceinline__ v16h load_a32(const TIN* __restrict__ p) {
  v16h a;
#pragma unroll
  for (int e = 0; e < 8; ++e) a[e] = (_Float16)p[e];
#pragma unroll
  for (int e = 0; e < 8; ++e) a[8 + e] = (_Float16)p[16 + e];
  return a;
}

// B-fragment (32x16): p points at (k0 + (l>>4)*8, col); elements are {p[0..7], p[16..23]}.
__device__ __forceinline__ v16h load_b_contig(const _Float16* __restrict__ p) {
  v16h b;
#pragma unroll
  for (int e = 0; e < 16; ++e) b[e] = p[(e & 7) + (e >> 3) * 16];
  return b;
}

// ---------------------------------------------------------------------------
// K0: weight transpose + f32->f16 convert.  W[k][n] -> WT[n][k].
// 32x32 tile through LDS; coalesced reads, 8B vector writes.
// ---------------------------------------------------------------------------
__global__ void __launch_bounds__(256)
wtrans_kernel(const float* __restrict__ W, _Float16* __restrict__ WT) {
  __shared__ float tile[32][33];
  const int t  = threadIdx.x;
  const int k0 = blockIdx.y * 32;
  const int n0 = blockIdx.x * 32;
  const int r  = t >> 3;        // 0..31
  const int c4 = (t & 7) * 4;   // 0..28

  const float4 v = *(const float4*)(W + (size_t)(k0 + r) * CH + n0 + c4);
  tile[r][c4 + 0] = v.x; tile[r][c4 + 1] = v.y;
  tile[r][c4 + 2] = v.z; tile[r][c4 + 3] = v.w;
  __syncthreads();

  v4h o;
#pragma unroll
  for (int e = 0; e < 4; ++e) o[e] = (_Float16)tile[c4 + e][r];
  *(v4h*)(WT + (size_t)(n0 + r) * CH + k0 + c4) = o;
}

// ---------------------------------------------------------------------------
// K1 / K4: GEMM  out = (X[4096x1024] @ W[1024x1024] + bias) * scale
// B comes from pre-transposed f16 WT[n][k] -> contiguous per-lane loads.
// Block = 256 thr = 8 waves; wave computes 32 rows x 64 cols = 8 WMMA tiles
// (2 A-frags reuse the same 4 B-frags each k-step).
// mode 0: f16 row-major   mode 1: f16 head-transposed (V)   mode 2: f32
// ---------------------------------------------------------------------------
template <typename TIN>
__global__ void __launch_bounds__(256)
gemm_bias_kernel(const TIN* __restrict__ X, const _Float16* __restrict__ WT,
                 const float* __restrict__ bias, void* __restrict__ out,
                 float scale, int mode) {
  const int lane = threadIdx.x & 31;
  const int w    = threadIdx.x >> 5;
  const int hm   = lane & 15;      // C/D col index, A row index
  const int hk   = lane >> 4;      // half-select
  const int m0   = blockIdx.y * 256 + w * 32;
  const int n0   = blockIdx.x * 64;

  v8f acc[2][4];
#pragma unroll
  for (int mi = 0; mi < 2; ++mi)
#pragma unroll
    for (int t = 0; t < 4; ++t)
      acc[mi][t] = v8f{0.f,0.f,0.f,0.f,0.f,0.f,0.f,0.f};

  const TIN* arow0 = X + (size_t)(m0 + hm) * CH + hk * 8;
  const TIN* arow1 = arow0 + (size_t)16 * CH;
  const _Float16* wtp = WT + (size_t)(n0 + hm) * CH + hk * 8;

  for (int k0 = 0; k0 < CH; k0 += 32) {
    __builtin_prefetch(arow0 + k0 + 128, 0, 1);   // next A chunk
    v16h a0 = load_a32(arow0 + k0);
    v16h a1 = load_a32(arow1 + k0);
#pragma unroll
    for (int t = 0; t < 4; ++t) {
      v16h b = load_b_contig(wtp + (size_t)(t * 16) * CH + k0);
      acc[0][t] = wmma16(a0, b, acc[0][t]);
      acc[1][t] = wmma16(a1, b, acc[1][t]);
    }
  }

#pragma unroll
  for (int mi = 0; mi < 2; ++mi) {
    const int mbase = m0 + mi * 16;
#pragma unroll
    for (int t = 0; t < 4; ++t) {
      const int n  = n0 + t * 16 + hm;
      const float bv = bias[n];
#pragma unroll
      for (int r = 0; r < 8; ++r) {
        const int m = mbase + r + hk * 8;
        const float v = (acc[mi][t][r] + bv) * scale;
        if (mode == 0) {
          ((_Float16*)out)[(size_t)m * CH + n] = (_Float16)v;
        } else if (mode == 1) {
          // head-transposed V: [bat][c][i]
          const int nb = m >> 11, li = m & 2047;
          const int a  = li >> 7;
          const int i  = ((li & 127) << 4) + (n >> 6);
          const int c  = n & 63;
          const int bat = nb * CHEADS + a;
          ((_Float16*)out)[((size_t)bat * CHD + c) * CL + i] = (_Float16)v;
        } else {
          ((float*)out)[(size_t)m * CH + n] = v;
        }
      }
    }
  }
}

// ---------------------------------------------------------------------------
// K2: column stats. Wave owns 16 key-columns j, scans query tiles i0 = j0..L.
// S tile = Q(16x64) . K(16x64)^T via 2 WMMA; online max/sum per column.
// ---------------------------------------------------------------------------
__global__ void __launch_bounds__(256)
colstats_kernel(const _Float16* __restrict__ qbuf,
                const _Float16* __restrict__ kbuf,
                float* __restrict__ mbuf, float* __restrict__ sbuf) {
  const int lane = threadIdx.x & 31;
  const int w    = threadIdx.x >> 5;
  const int hm   = lane & 15, hk = lane >> 4;
  const int bat  = blockIdx.y;
  const int j0   = blockIdx.x * 128 + w * 16;

  const _Float16* qb = qbuf + (size_t)bat * CL * CHD;
  const _Float16* kb = kbuf + (size_t)bat * CL * CHD;

  // K^T B-fragments for this column tile (k-dim = c, table order per lane)
  const _Float16* kp = kb + (size_t)(j0 + hm) * CHD + hk * 8;
  const v16h bk0 = load_b_contig(kp);        // c = 0..31
  const v16h bk1 = load_b_contig(kp + 32);   // c = 32..63

  float M = -3.0e38f, S = 0.0f;
  const int j = j0 + hm;

  for (int i0 = j0; i0 < CL; i0 += 16) {
    const _Float16* qp = qb + (size_t)(i0 + hm) * CHD + hk * 8;
    v16h a0 = load_a32(qp);
    v16h a1 = load_a32(qp + 32);
    v8f s = v8f{0.f,0.f,0.f,0.f,0.f,0.f,0.f,0.f};
    s = wmma16(a0, bk0, s);
    s = wmma16(a1, bk1, s);

    float vals[8], cm = -3.0e38f;
#pragma unroll
    for (int r = 0; r < 8; ++r) {
      const int i = i0 + r + hk * 8;
      vals[r] = (i >= j) ? s[r] : -3.0e38f;
      cm = fmaxf(cm, vals[r]);
    }
    const float Mn = fmaxf(M, cm);
    S *= __expf(M - Mn);
#pragma unroll
    for (int r = 0; r < 8; ++r) S += __expf(vals[r] - Mn);
    M = Mn;
  }

  // merge the two half-wave partials (rows r and r+8 live in lane pairs l,l+16)
  const float Mo = __shfl_xor(M, 16, 32);
  const float So = __shfl_xor(S, 16, 32);
  const float Mf = fmaxf(M, Mo);
  const float Sf = S * __expf(M - Mf) + So * __expf(Mo - Mf);
  if (lane < 16) {
    mbuf[(size_t)bat * CL + j] = Mf;
    sbuf[(size_t)bat * CL + j] = Sf;
  }
}

// ---------------------------------------------------------------------------
// K3: context. Wave owns a 16-query tile i0; loops causal j-chunks of 32.
// Recomputes S, forms P = exp(S-m_j)/s_j, stages P in LDS (C-layout ->
// A-layout relayout), then 4 WMMA vs head-transposed V.
// ---------------------------------------------------------------------------
__global__ void __launch_bounds__(256)
context_kernel(const _Float16* __restrict__ qbuf,
               const _Float16* __restrict__ kbuf,
               const _Float16* __restrict__ vT,
               const float* __restrict__ mbuf, const float* __restrict__ sbuf,
               _Float16* __restrict__ cbuf) {
  __shared__ _Float16 pst[8][16 * 40];   // per-wave 16x32 P tile, stride 40 halves

  const int lane = threadIdx.x & 31;
  const int w    = threadIdx.x >> 5;
  const int hm   = lane & 15, hk = lane >> 4;
  const int bat  = blockIdx.y;
  const int i0   = blockIdx.x * 128 + w * 16;

  const _Float16* qb  = qbuf + (size_t)bat * CL * CHD;
  const _Float16* kb  = kbuf + (size_t)bat * CL * CHD;
  const _Float16* vb  = vT   + (size_t)bat * CHD * CL;
  const float* mrow   = mbuf + (size_t)bat * CL;
  const float* srow   = sbuf + (size_t)bat * CL;

  // Q A-fragments for this query tile (invariant over j)
  const _Float16* qp = qb + (size_t)(i0 + hm) * CHD + hk * 8;
  const v16h aq0 = load_a32(qp);
  const v16h aq1 = load_a32(qp + 32);

  v8f acc[4];
#pragma unroll
  for (int t = 0; t < 4; ++t) acc[t] = v8f{0.f,0.f,0.f,0.f,0.f,0.f,0.f,0.f};

  _Float16* myp = &pst[w][0];

  for (int j0 = 0; j0 <= i0; j0 += 32) {
#pragma unroll
    for (int sub = 0; sub < 2; ++sub) {
      const int js = j0 + sub * 16;
      if (js <= i0) {
        const _Float16* kp = kb + (size_t)(js + hm) * CHD + hk * 8;
        v16h b0 = load_b_contig(kp);
        v16h b1 = load_b_contig(kp + 32);
        v8f s = v8f{0.f,0.f,0.f,0.f,0.f,0.f,0.f,0.f};
        s = wmma16(aq0, b0, s);
        s = wmma16(aq1, b1, s);

        const int j  = js + hm;
        const float mj = mrow[j];
        const float rs = 1.0f / srow[j];
#pragma unroll
        for (int r = 0; r < 8; ++r) {
          const int i = i0 + r + hk * 8;
          const float p = (i >= j) ? __expf(s[r] - mj) * rs : 0.0f;
          myp[(r + hk * 8) * 40 + sub * 16 + hm] = (_Float16)p;
        }
      } else {
#pragma unroll
        for (int r = 0; r < 8; ++r)
          myp[(r + hk * 8) * 40 + sub * 16 + hm] = (_Float16)0.0f;
      }
    }
    // CDNA5 split counter: make the per-wave LDS stores visible to our loads
    asm volatile("s_wait_dscnt 0" ::: "memory");

    // P A-fragment (rows = i, k = local j 0..31)
    v16h pa;
    {
      const _Float16* pp = myp + hm * 40 + hk * 8;
#pragma unroll
      for (int e = 0; e < 8; ++e) pa[e] = pp[e];
#pragma unroll
      for (int e = 0; e < 8; ++e) pa[8 + e] = pp[16 + e];
    }
#pragma unroll
    for (int t = 0; t < 4; ++t) {
      const _Float16* vp = vb + (size_t)(t * 16 + hm) * CL + j0 + hk * 8;
      v16h bvf = load_b_contig(vp);
      acc[t] = wmma16(pa, bvf, acc[t]);
    }
  }

#pragma unroll
  for (int t = 0; t < 4; ++t)
#pragma unroll
    for (int r = 0; r < 8; ++r) {
      const int i = i0 + r + hk * 8;
      const int c = t * 16 + hm;
      cbuf[((size_t)bat * CL + i) * CHD + c] = (_Float16)acc[t][r];
    }
}

// ---------------------------------------------------------------------------
extern "C" void kernel_launch(void* const* d_in, const int* in_sizes, int n_in,
                              void* d_out, int out_size, void* d_ws, size_t ws_size,
                              hipStream_t stream) {
  (void)in_sizes; (void)n_in; (void)out_size; (void)ws_size;
  const float* Hq = (const float*)d_in[0];
  const float* Hk = (const float*)d_in[1];
  const float* Hv = (const float*)d_in[2];
  const float* Wq = (const float*)d_in[3];
  const float* bq = (const float*)d_in[4];
  const float* Wk = (const float*)d_in[5];
  const float* bk = (const float*)d_in[6];
  const float* Wv = (const float*)d_in[7];
  const float* bv = (const float*)d_in[8];
  const float* Wo = (const float*)d_in[9];
  const float* bo = (const float*)d_in[10];
  // d_in[11] = mask (causal, handled analytically)

  // workspace carve-up (~40.5 MB)
  const size_t nelem = (size_t)MROWS * CH;       // 4096*1024
  const size_t welem = (size_t)CH * CH;          // 1024*1024
  _Float16* qbuf = (_Float16*)d_ws;
  _Float16* kbuf = qbuf + nelem;
  _Float16* vTb  = kbuf + nelem;
  _Float16* cbuf = vTb  + nelem;
  _Float16* wqT  = cbuf + nelem;
  _Float16* wkT  = wqT  + welem;
  _Float16* wvT  = wkT  + welem;
  _Float16* woT  = wvT  + welem;
  float*    mbuf = (float*)(woT + welem);
  float*    sbuf = mbuf + (size_t)NBAT * CL;

  const float scale = 1.0f / 32.0f;              // 1/sqrt(H)

  dim3 blk(256);
  dim3 gT(CH / 32, CH / 32);                     // (32, 32) transpose tiles
  dim3 gG(CH / 64, MROWS / 256);                 // (16, 16) GEMM tiles
  dim3 gA(CL / 128, NBAT);                       // (16, 32) attention tiles

  wtrans_kernel<<<gT, blk, 0, stream>>>(Wq, wqT);
  wtrans_kernel<<<gT, blk, 0, stream>>>(Wk, wkT);
  wtrans_kernel<<<gT, blk, 0, stream>>>(Wv, wvT);
  wtrans_kernel<<<gT, blk, 0, stream>>>(Wo, woT);

  gemm_bias_kernel<float><<<gG, blk, 0, stream>>>(Hq, wqT, bq, qbuf, scale, 0);
  gemm_bias_kernel<float><<<gG, blk, 0, stream>>>(Hk, wkT, bk, kbuf, scale, 0);
  gemm_bias_kernel<float><<<gG, blk, 0, stream>>>(Hv, wvT, bv, vTb,  1.0f,  1);

  colstats_kernel<<<gA, blk, 0, stream>>>(qbuf, kbuf, mbuf, sbuf);
  context_kernel<<<gA, blk, 0, stream>>>(qbuf, kbuf, vTb, mbuf, sbuf, cbuf);

  gemm_bias_kernel<_Float16><<<gG, blk, 0, stream>>>(cbuf, woT, bo, (float*)d_out,
                                                     1.0f, 2);
}
